// DifferentiablePooling_57775900066445
// MI455X (gfx1250) — compile-verified, matrix operand order checked by reading, not run on hardware
//
#include <hip/hip_runtime.h>
#include <hip/hip_bf16.h>

typedef __attribute__((ext_vector_type(16))) __bf16 v16bf;
typedef __attribute__((ext_vector_type(8)))  float  v8f;

#define N_NODES 50000
#define N_EDGES 800000
#define DIN 512
#define DH  256
#define KMAX 50
#define SPAD 64   // padded column count for layer-2 (4 WMMA n-tiles)

// ---------------------------------------------------------------------------
// Degree / normalization
// ---------------------------------------------------------------------------
__global__ void init_deg_kernel(float* __restrict__ deg, float* __restrict__ loss) {
    int i = blockIdx.x * blockDim.x + threadIdx.x;
    if (i < N_NODES) deg[i] = 1.0f;   // self-loop
    if (i == 0) loss[0] = 0.0f;
}

__global__ void deg_edge_kernel(const long long* __restrict__ dst, float* __restrict__ deg) {
    int e = blockIdx.x * blockDim.x + threadIdx.x;
    if (e < N_EDGES) atomicAdd(&deg[(int)dst[e]], 1.0f);
}

__global__ void to_dinv_kernel(float* __restrict__ deg) {
    int i = blockIdx.x * blockDim.x + threadIdx.x;
    if (i < N_NODES) deg[i] = rsqrtf(deg[i]);   // deg >= 1 always
}

// ---------------------------------------------------------------------------
// GEMM1: H = X(N,512) @ W1(512,256), bf16 WMMA, f32 accum.
// grid.x = N/16 blocks, 256 threads (8 waves); wave w -> n-tiles {2w, 2w+1}.
// ---------------------------------------------------------------------------
__global__ __launch_bounds__(256)
void gemm1_kernel(const float* __restrict__ X, const float* __restrict__ W1,
                  float* __restrict__ H) {
    const int lane   = threadIdx.x & 31;
    const int wave   = threadIdx.x >> 5;
    const int m_base = blockIdx.x << 4;
    const int mrow   = lane & 15;
    const int kshift = (lane >> 4) << 3;   // 0 or 8 (K sub-offset per half-wave)
    const int n0     = wave << 5;          // two 16-wide n-tiles per wave

    v8f acc0 = {}; v8f acc1 = {};
    const float* xr = X  + (size_t)(m_base + mrow) * DIN + kshift;
    const float* wb = W1 + (size_t)kshift * DH + n0 + mrow;

    for (int kb = 0; kb < DIN; kb += 32) {
        v16bf a, b0, b1v;
        #pragma unroll
        for (int i = 0; i < 8; ++i) {
            a[i]     = (__bf16)xr[kb + i];        // K = kb+kshift+i
            a[8 + i] = (__bf16)xr[kb + 16 + i];   // K = kb+kshift+16+i
        }
        const float* wp = wb + (size_t)kb * DH;
        #pragma unroll
        for (int i = 0; i < 8; ++i) {
            b0[i]      = (__bf16)wp[(size_t)i * DH];
            b0[8 + i]  = (__bf16)wp[(size_t)(16 + i) * DH];
            b1v[i]     = (__bf16)wp[(size_t)i * DH + 16];
            b1v[8 + i] = (__bf16)wp[(size_t)(16 + i) * DH + 16];
        }
        acc0 = __builtin_amdgcn_wmma_f32_16x16x32_bf16(false, a, false, b0,  (short)0, acc0, false, false);
        acc1 = __builtin_amdgcn_wmma_f32_16x16x32_bf16(false, a, false, b1v, (short)0, acc1, false, false);
    }
    const int mo = (lane >> 4) << 3;   // C/D: M = r + 8*(lane>=16), N = lane&15
    float* out = H + (size_t)(m_base + mo) * DH + n0 + mrow;
    #pragma unroll
    for (int r = 0; r < 8; ++r) {
        out[(size_t)r * DH]      = acc0[r];
        out[(size_t)r * DH + 16] = acc1[r];
    }
}

// ---------------------------------------------------------------------------
// GEMM2: Sraw = h(N,256) @ W2(256,50) into padded (N,64).
// grid.x = N/16 blocks, 128 threads (4 waves); wave w -> n-tile w.
// ---------------------------------------------------------------------------
__global__ __launch_bounds__(128)
void gemm2_kernel(const float* __restrict__ Hm, const float* __restrict__ W2,
                  float* __restrict__ Sraw) {
    const int lane   = threadIdx.x & 31;
    const int wave   = threadIdx.x >> 5;
    const int m_base = blockIdx.x << 4;
    const int mrow   = lane & 15;
    const int kshift = (lane >> 4) << 3;
    const int n      = (wave << 4) + mrow;

    v8f acc = {};
    const float* hr = Hm + (size_t)(m_base + mrow) * DH + kshift;

    for (int kb = 0; kb < DH; kb += 32) {
        v16bf a, b;
        #pragma unroll
        for (int i = 0; i < 8; ++i) {
            a[i]     = (__bf16)hr[kb + i];
            a[8 + i] = (__bf16)hr[kb + 16 + i];
        }
        #pragma unroll
        for (int i = 0; i < 8; ++i) {
            const int k0 = kb + kshift + i;
            const int k1 = kb + kshift + 16 + i;
            b[i]     = (n < KMAX) ? (__bf16)W2[(size_t)k0 * KMAX + n] : (__bf16)0.0f;
            b[8 + i] = (n < KMAX) ? (__bf16)W2[(size_t)k1 * KMAX + n] : (__bf16)0.0f;
        }
        acc = __builtin_amdgcn_wmma_f32_16x16x32_bf16(false, a, false, b, (short)0, acc, false, false);
    }
    const int mo = (lane >> 4) << 3;
    float* out = Sraw + (size_t)(m_base + mo) * SPAD + (wave << 4) + mrow;
    #pragma unroll
    for (int r = 0; r < 8; ++r) out[(size_t)r * SPAD] = acc[r];
}

// ---------------------------------------------------------------------------
// GCN propagation: self-loop term (plain store) + edge scatter (atomics).
// ldshift = log2(row stride): 8 for layer1 (256), 6 for layer2 (64).
// ---------------------------------------------------------------------------
__global__ void self_scale_kernel(const float* __restrict__ h, const float* __restrict__ dinv,
                                  float* __restrict__ agg, int ldshift, long long total) {
    long long t = (long long)blockIdx.x * blockDim.x + threadIdx.x;
    if (t < total) {
        int i = (int)(t >> ldshift);
        float dv = dinv[i];
        agg[t] = dv * dv * h[t];
    }
}

__global__ void edge_scatter_kernel(const float* __restrict__ h, const float* __restrict__ dinv,
                                    const long long* __restrict__ src, const long long* __restrict__ dst,
                                    float* __restrict__ agg, int ldshift) {
    const int gshift = ldshift - 2;                      // channel groups of 4
    long long t = (long long)blockIdx.x * blockDim.x + threadIdx.x;
    long long e = t >> gshift;
    if (e < N_EDGES) {
        int c  = (int)(t & ((1 << gshift) - 1)) << 2;
        int si = (int)src[e], di = (int)dst[e];
        float norm = dinv[si] * dinv[di];
        const float4 v = *reinterpret_cast<const float4*>(h + ((size_t)si << ldshift) + c);
        float* o = agg + ((size_t)di << ldshift) + c;
        atomicAdd(o + 0, norm * v.x);
        atomicAdd(o + 1, norm * v.y);
        atomicAdd(o + 2, norm * v.z);
        atomicAdd(o + 3, norm * v.w);
    }
}

__global__ void relu_bias_kernel(float* __restrict__ h, const float* __restrict__ agg,
                                 const float* __restrict__ b1) {
    long long t = (long long)blockIdx.x * blockDim.x + threadIdx.x;
    if (t < (long long)N_NODES * DH) {
        int c = (int)(t & (DH - 1));
        h[t] = fmaxf(agg[t] + b1[c], 0.0f);
    }
}

// ---------------------------------------------------------------------------
// Row softmax over 50 classes: one wave per node (wave32).
// ---------------------------------------------------------------------------
__global__ __launch_bounds__(256)
void softmax_kernel(const float* __restrict__ agg2, const float* __restrict__ b2,
                    float* __restrict__ S) {
    const int lane = threadIdx.x & 31;
    const int wave = threadIdx.x >> 5;
    const int node = blockIdx.x * 8 + wave;
    if (node >= N_NODES) return;
    const float* row = agg2 + (size_t)node * SPAD;
    float v1 = row[lane] + b2[lane];                    // lane < 32 < 50 always valid
    float v2 = (lane + 32 < KMAX) ? row[lane + 32] + b2[lane + 32] : -INFINITY;
    float m = fmaxf(v1, v2);
    #pragma unroll
    for (int off = 16; off; off >>= 1) m = fmaxf(m, __shfl_xor(m, off, 32));
    float e1 = __expf(v1 - m);
    float e2 = (lane + 32 < KMAX) ? __expf(v2 - m) : 0.0f;
    float s = e1 + e2;
    #pragma unroll
    for (int off = 16; off; off >>= 1) s += __shfl_xor(s, off, 32);
    float inv = 1.0f / s;
    S[(size_t)node * KMAX + lane] = e1 * inv;
    if (lane + 32 < KMAX) S[(size_t)node * KMAX + lane + 32] = e2 * inv;
}

// ---------------------------------------------------------------------------
// Spatial loss: one wave per edge, block-reduced, one atomic per block.
// ---------------------------------------------------------------------------
__global__ __launch_bounds__(256)
void spatial_kernel(const float* __restrict__ S, const float* __restrict__ pos,
                    const long long* __restrict__ src, const long long* __restrict__ dst,
                    float* __restrict__ loss) {
    __shared__ float part[8];
    const int lane = threadIdx.x & 31;
    const int wave = threadIdx.x >> 5;
    const long long e = (long long)blockIdx.x * 8 + wave;
    float contrib = 0.0f;
    if (e < N_EDGES) {
        int si = (int)src[e], di = (int)dst[e];
        float dx = pos[2 * si]     - pos[2 * di];
        float dy = pos[2 * si + 1] - pos[2 * di + 1];
        float d2 = dx * dx + dy * dy;
        const float* Ss = S + (size_t)si * KMAX;
        const float* Sd = S + (size_t)di * KMAX;
        float dot = Ss[lane] * Sd[lane];
        if (lane + 32 < KMAX) dot += Ss[lane + 32] * Sd[lane + 32];
        #pragma unroll
        for (int off = 16; off; off >>= 1) dot += __shfl_xor(dot, off, 32);
        contrib = d2 * dot;
    }
    if (lane == 0) part[wave] = contrib;
    __syncthreads();
    if (threadIdx.x == 0) {
        float t = 0.0f;
        #pragma unroll
        for (int i = 0; i < 8; ++i) t += part[i];
        atomicAdd(loss, t);
    }
}

__global__ void finalize_kernel(const float* __restrict__ loss, const float* __restrict__ sw,
                                float* __restrict__ out) {
    out[0] = loss[0] * sw[0] * (1.0f / (float)N_EDGES);
}

// ---------------------------------------------------------------------------
extern "C" void kernel_launch(void* const* d_in, const int* in_sizes, int n_in,
                              void* d_out, int out_size, void* d_ws, size_t ws_size,
                              hipStream_t stream) {
    const float*     X   = (const float*)d_in[0];
    const float*     pos = (const float*)d_in[1];
    const float*     W1  = (const float*)d_in[2];
    const float*     b1  = (const float*)d_in[3];
    const float*     W2  = (const float*)d_in[4];
    const float*     b2  = (const float*)d_in[5];
    const float*     sw  = (const float*)d_in[6];
    const long long* ei  = (const long long*)d_in[7];   // int64 edge_index (2,E)
    const long long* src = ei;
    const long long* dst = ei + N_EDGES;

    float* S_out = (float*)d_out;                        // (N, 50) then scalar loss

    // Workspace layout (floats)
    float* ws    = (float*)d_ws;
    float* dinv  = ws;                                   // N        (deg -> dinv in place)
    float* h     = dinv + N_NODES;                       // N*256    (X@W1, then relu'd h)
    float* agg1  = h + (size_t)N_NODES * DH;             // N*256    (layer-1 aggregate)
    float* sbuf  = agg1;                                 // N*64     (reuse after relu)
    float* agg2  = agg1 + (size_t)N_NODES * SPAD;        // N*64     (layer-2 aggregate)
    float* loss  = agg1 + (size_t)N_NODES * DH;          // 1        (after agg1 region)

    const int TB = 256;

    // degrees + normalization
    init_deg_kernel<<<(N_NODES + TB - 1) / TB, TB, 0, stream>>>(dinv, loss);
    deg_edge_kernel<<<N_EDGES / TB, TB, 0, stream>>>(dst, dinv);
    to_dinv_kernel<<<(N_NODES + TB - 1) / TB, TB, 0, stream>>>(dinv);

    // layer 1
    gemm1_kernel<<<N_NODES / 16, 256, 0, stream>>>(X, W1, h);
    self_scale_kernel<<<(int)(((long long)N_NODES * DH) / TB), TB, 0, stream>>>(
        h, dinv, agg1, 8, (long long)N_NODES * DH);
    edge_scatter_kernel<<<(int)(((long long)N_EDGES * (DH / 4)) / TB), TB, 0, stream>>>(
        h, dinv, src, dst, agg1, 8);
    relu_bias_kernel<<<(int)(((long long)N_NODES * DH) / TB), TB, 0, stream>>>(h, agg1, b1);

    // layer 2
    gemm2_kernel<<<N_NODES / 16, 128, 0, stream>>>(h, W2, sbuf);
    self_scale_kernel<<<(int)(((long long)N_NODES * SPAD) / TB), TB, 0, stream>>>(
        sbuf, dinv, agg2, 6, (long long)N_NODES * SPAD);
    edge_scatter_kernel<<<(int)(((long long)N_EDGES * (SPAD / 4)) / TB), TB, 0, stream>>>(
        sbuf, dinv, src, dst, agg2, 6);

    // softmax + spatial loss
    softmax_kernel<<<N_NODES / 8, 256, 0, stream>>>(agg2, b2, S_out);
    spatial_kernel<<<N_EDGES / 8, 256, 0, stream>>>(S_out, pos, src, dst, loss);
    finalize_kernel<<<1, 1, 0, stream>>>(loss, sw, S_out + (size_t)N_NODES * KMAX);

    (void)in_sizes; (void)n_in; (void)out_size; (void)ws_size;
}
